// MultiLayerDeepSNN_40750649705153
// MI455X (gfx1250) — matrix operand order repaired and checked by reading, not compiled, not run on hardware
//
#include <hip/hip_runtime.h>

// CDNA5 (gfx1250) fused spiking-MLP: 15 timesteps x 6 layers in one persistent
// kernel. Membrane states live in WMMA accumulator fragments (VGPRs), spikes
// ping-pong through LDS, weights are f16 (exact 0/1 spikes => f16 A exact).
// 512 threads = 16 wave32; each wave owns 2 N-tiles. A per-step asm memory
// clobber stops LICM from hoisting (and then spilling) the loop-invariant
// weight loads; B-fragments are software-pipelined one K-tile ahead.

typedef __attribute__((ext_vector_type(16))) _Float16 v16h;
typedef __attribute__((ext_vector_type(8)))  _Float16 v8h;
typedef __attribute__((ext_vector_type(8)))  float    v8f;

#define THRV   1.0f
#define RS     264      // LDS spike-row stride in halves (256 + 8 pad -> bank rotate)
#define NSTEPS 15

union AFrag { v16h v; v8h h[2]; };

// Offsets (in halves) of f16 weights inside d_ws
#define OW2 0
#define OW3 65536
#define OW4 131072
#define OW5 196608
#define WTOT 229376     // 3*256*256 + 128*256 halves

__global__ void snn_prep_weights(const float* __restrict__ W2,
                                 const float* __restrict__ W3,
                                 const float* __restrict__ W4,
                                 const float* __restrict__ W5,
                                 _Float16* __restrict__ out) {
    int i = blockIdx.x * blockDim.x + threadIdx.x;
    if (i < OW3)            out[i] = (_Float16)W2[i - OW2];
    else if (i < OW4)       out[i] = (_Float16)W3[i - OW3];
    else if (i < OW5)       out[i] = (_Float16)W4[i - OW4];
    else if (i < WTOT)      out[i] = (_Float16)W5[i - OW5];
}

// One LIF layer: cur = spk_in @ W.T + b (WMMA, K=256), then membrane update
// and spike write. NT = number of 16-wide N tiles owned by this wave.
template<int NT>
__device__ __forceinline__ void lif_layer(
    const _Float16* __restrict__ Wl,   // [N][256] f16 row-major (W.T(k,n)=Wl[n*256+k])
    const float* __restrict__ bl,      // [N] f32 bias
    const _Float16* bufIn,             // LDS spikes in  [32][RS]
    _Float16* bufOut,                  // LDS spikes out [32][RS]
    float beta, v8f (&mst)[NT],
    int mgrp, int nbase, int half, int l15)
{
    v8f c[NT];
    const _Float16* brow[NT];
    #pragma unroll
    for (int t = 0; t < NT; ++t) {
        const int n = nbase + t * 16 + l15;
        const float bn = bl[n];
        #pragma unroll
        for (int r = 0; r < 8; ++r) c[t][r] = bn;
        brow[t] = Wl + n * 256 + half * 16;    // B frag base for this lane/tile
    }

    const int arowOff = (mgrp * 16 + l15) * RS;   // A: lane M = lane&15

    // Software-pipelined B fragments: one K-tile of lookahead.
    v16h bcur[NT];
    #pragma unroll
    for (int t = 0; t < NT; ++t) bcur[t] = *(const v16h*)(brow[t]);

    #pragma unroll
    for (int kt = 0; kt < 8; ++kt) {              // K = 256 = 8 x 32
        // A fragment: elems 0-15 -> K=half*16+i (matches B contiguous order)
        // => two 16B LDS loads.
        AFrag a;
        const v8h* ap = (const v8h*)(bufIn + arowOff + kt * 32 + half * 16);
        a.h[0] = ap[0];
        a.h[1] = ap[1];

        v16h bnxt[NT];
        if (kt < 7) {
            #pragma unroll
            for (int t = 0; t < NT; ++t) {
                bnxt[t] = *(const v16h*)(brow[t] + (kt + 1) * 32);
                if (kt < 6) __builtin_prefetch(brow[t] + (kt + 2) * 32, 0, 3);
            }
        }
        #pragma unroll
        for (int t = 0; t < NT; ++t)
            c[t] = __builtin_amdgcn_wmma_f32_16x16x32_f16(
                false, a.v, false, bcur[t], (short)0, c[t], false, false);
        if (kt < 7) {
            #pragma unroll
            for (int t = 0; t < NT; ++t) bcur[t] = bnxt[t];
        }
    }

    // LIF: reset from PREVIOUS mem; spike = heaviside(mem_new - 1)
    const int orow = mgrp * 16 + half * 8;        // C/D: lane N = lane&15, M = half*8+r
    #pragma unroll
    for (int t = 0; t < NT; ++t) {
        const int col = nbase + t * 16 + l15;
        #pragma unroll
        for (int r = 0; r < 8; ++r) {
            const float mo = mst[t][r];
            const float mn = beta * mo + c[t][r] - ((mo > THRV) ? THRV : 0.0f);
            mst[t][r] = mn;
            bufOut[(orow + r) * RS + col] = (_Float16)((mn > THRV) ? 1.0f : 0.0f);
        }
    }
}

__global__ __launch_bounds__(512)
void snn_fused(const float* __restrict__ x,
               const float* __restrict__ W1, const float* __restrict__ b1,
               const float* __restrict__ b2, const float* __restrict__ b3,
               const float* __restrict__ b4, const float* __restrict__ b5,
               const float* __restrict__ W6, const float* __restrict__ b6,
               const _Float16* __restrict__ Wh,
               float* __restrict__ out)
{
    __shared__ __align__(16) _Float16 bufA[32 * RS];
    __shared__ __align__(16) _Float16 bufB[32 * RS];
    __shared__ float xs[32 * 6];
    __shared__ float W6s[3 * 128];
    __shared__ float b6s[3];
    __shared__ float accs[32 * 3];

    const int tid  = threadIdx.x;
    const int lane = tid & 31;
    const int wave = tid >> 5;         // 0..15
    const int mgrp = wave >> 3;        // which 16-row block of the 32-row tile
    const int ng   = wave & 7;         // n-tile group (8 groups x 2 tiles = 256)
    const int half = lane >> 4;
    const int l15  = lane & 15;
    const int wgBase = blockIdx.x * 32;

    for (int i = tid; i < 32 * 6; i += 512) xs[i] = x[wgBase * 6 + i];
    if (tid < 3 * 128) W6s[tid] = W6[tid];
    if (tid < 3)  b6s[tid] = b6[tid];
    if (tid < 96) accs[tid] = 0.0f;
    __syncthreads();

    // cur1 = x @ W1.T + b1  (K=6, constant over timesteps) in C/D layout
    v8f c1[2];
    {
        const int mrow0 = mgrp * 16 + half * 8;
        #pragma unroll
        for (int t = 0; t < 2; ++t) {
            const int n = ng * 32 + t * 16 + l15;
            float w1r[6];
            #pragma unroll
            for (int k = 0; k < 6; ++k) w1r[k] = W1[n * 6 + k];
            const float bn = b1[n];
            #pragma unroll
            for (int r = 0; r < 8; ++r) {
                float acc = bn;
                #pragma unroll
                for (int k = 0; k < 6; ++k) acc += xs[(mrow0 + r) * 6 + k] * w1r[k];
                c1[t][r] = acc;
            }
        }
    }

    v8f m1[2], m2[2], m3[2], m4[2], m5[1];
    #pragma unroll
    for (int t = 0; t < 2; ++t)
        #pragma unroll
        for (int r = 0; r < 8; ++r) { m1[t][r] = 0.f; m2[t][r] = 0.f; m3[t][r] = 0.f; m4[t][r] = 0.f; }
    #pragma unroll
    for (int r = 0; r < 8; ++r) m5[0][r] = 0.f;

    #pragma unroll 1
    for (int s = 0; s < NSTEPS; ++s) {
        // Stop LICM from hoisting the (loop-invariant) weight loads out of the
        // step loop -- that caused global->scratch spill/reload of every B frag.
        asm volatile("" ::: "memory");

        // ---- layer 1 (constant input current) -> spikes in bufA ----
        {
            const int orow = mgrp * 16 + half * 8;
            #pragma unroll
            for (int t = 0; t < 2; ++t) {
                const int col = ng * 32 + t * 16 + l15;
                #pragma unroll
                for (int r = 0; r < 8; ++r) {
                    const float mo = m1[t][r];
                    const float mn = 0.90f * mo + c1[t][r] - ((mo > THRV) ? THRV : 0.0f);
                    m1[t][r] = mn;
                    bufA[(orow + r) * RS + col] = (_Float16)((mn > THRV) ? 1.0f : 0.0f);
                }
            }
        }
        __syncthreads();
        lif_layer<2>(Wh + OW2, b2, bufA, bufB, 0.88f, m2, mgrp, ng * 32, half, l15);
        __syncthreads();
        lif_layer<2>(Wh + OW3, b3, bufB, bufA, 0.86f, m3, mgrp, ng * 32, half, l15);
        __syncthreads();
        lif_layer<2>(Wh + OW4, b4, bufA, bufB, 0.84f, m4, mgrp, ng * 32, half, l15);
        __syncthreads();
        lif_layer<1>(Wh + OW5, b5, bufB, bufA, 0.82f, m5, mgrp, ng * 16, half, l15);
        __syncthreads();
        // ---- layer 6: out += spk5 @ W6.T + b6 (3-wide, plain FMA) ----
        if (tid < 96) {
            const int m = tid / 3, j = tid % 3;
            float acc = b6s[j];
            #pragma unroll 8
            for (int k = 0; k < 128; ++k)
                acc += (float)bufA[m * RS + k] * W6s[j * 128 + k];
            accs[tid] += acc;
        }
        __syncthreads();
    }

    if (tid < 96)
        out[(wgBase + tid / 3) * 3 + (tid % 3)] = accs[tid] * (1.0f / (float)NSTEPS);
}

extern "C" void kernel_launch(void* const* d_in, const int* in_sizes, int n_in,
                              void* d_out, int out_size, void* d_ws, size_t ws_size,
                              hipStream_t stream) {
    const float* x  = (const float*)d_in[0];
    const float* W1 = (const float*)d_in[1];
    const float* b1 = (const float*)d_in[2];
    const float* W2 = (const float*)d_in[3];
    const float* b2 = (const float*)d_in[4];
    const float* W3 = (const float*)d_in[5];
    const float* b3 = (const float*)d_in[6];
    const float* W4 = (const float*)d_in[7];
    const float* b4 = (const float*)d_in[8];
    const float* W5 = (const float*)d_in[9];
    const float* b5 = (const float*)d_in[10];
    const float* W6 = (const float*)d_in[11];
    const float* b6 = (const float*)d_in[12];
    float* out = (float*)d_out;
    _Float16* Wh = (_Float16*)d_ws;     // 458752 B of f16 weights

    const int B = in_sizes[0] / 6;      // 16384

    snn_prep_weights<<<WTOT / 256, 256, 0, stream>>>(W2, W3, W4, W5, Wh);
    snn_fused<<<B / 32, 512, 0, stream>>>(x, W1, b1, b2, b3, b4, b5, W6, b6, Wh, out);
    (void)n_in; (void)out_size; (void)ws_size;
}
